// Decoder1_9010841387469
// MI455X (gfx1250) — matrix operand order repaired and checked by reading, not compiled, not run on hardware
//
#include <hip/hip_runtime.h>
#include <hip/hip_bf16.h>
#include <math.h>

// Problem constants (fixed by the reference).
#define BATCH 512
#define NNODE 101
#define DIM   128
#define NH    8
#define HD    16
#define NSTEP 140

typedef __attribute__((ext_vector_type(16))) _Float16 v16h;
typedef __attribute__((ext_vector_type(8)))  float    v8f;
typedef __attribute__((ext_vector_type(4)))  unsigned int u32x4;
typedef __attribute__((ext_vector_type(8)))  int      i32x8;
typedef __attribute__((ext_vector_type(4)))  int      i32x4;

#if __has_builtin(__builtin_amdgcn_tensor_load_to_lds) && \
    __has_builtin(__builtin_amdgcn_s_wait_tensorcnt)
#define USE_TDM 1
#endif

// ---------------------------------------------------------------------------
// WMMA helpers (CDNA5 v_wmma_f32_16x16x32_f16, wave32)
// ---------------------------------------------------------------------------
__device__ __forceinline__ v8f wmma16(v16h a, v16h b, v8f c) {
  // 8 args: (neg_a, A, neg_b, B, c_mod, C, reuse_a, reuse_b)
  return __builtin_amdgcn_wmma_f32_16x16x32_f16(false, a, false, b, (short)0, c,
                                                false, false);
}

// A fragment: 16x32 f16 tile from LDS, row-major [16][lda].
// Layout: lanes 0-15 hold M=0..15 with K 0..15 contiguous;
// lanes 16-31 same M, K 16..31 (matches ldB).
__device__ __forceinline__ v16h ldA(const _Float16* Ab, int lda, int k0, int lane) {
  int m  = lane & 15;
  int ks = (lane >> 4) * 16;
  const _Float16* p = Ab + m * lda + k0 + ks;
  v16h a;
#pragma unroll
  for (int r = 0; r < 4; ++r) { a[2*r]   = p[2*r];      a[2*r+1] = p[2*r+1]; }
#pragma unroll
  for (int r = 0; r < 4; ++r) { a[8+2*r] = p[8+2*r];    a[9+2*r] = p[8+2*r+1]; }
  return a;
}

// B fragment: 32x16 f16 tile. Weights stored TRANSPOSED in LDS: Bt[n][k], so each
// lane (column n = n0 + lane&15) reads 16 consecutive K values
// (lanes 0-15: K k0..k0+15, lanes 16-31: K k0+16..k0+31).
__device__ __forceinline__ v16h ldB(const _Float16* Bt, int ldb, int k0, int n0, int lane) {
  int n  = n0 + (lane & 15);
  int kb = k0 + (lane >> 4) * 16;
  const _Float16* p = Bt + n * ldb + kb;
  v16h b;
#pragma unroll
  for (int i = 0; i < 16; ++i) b[i] = p[i];
  return b;
}

#ifdef USE_TDM
// Issue a TDM async load of a [16 rows x 128 cols] f32 tile of the flat
// [51712,128] tensor into LDS at byte offset ldsOff.  D# per
// CDNA5 ISA (group0 128b, group1 256b, groups 2/3 zero:
// 2-D tile).  6-arg builtin form (amdgpu-toolchain / clang-23).
__device__ __forceinline__ void tdm_load_tile(unsigned ldsOff,
                                              unsigned long long gaddr) {
  u32x4 g0;
  g0[0] = 1u;                                               // count=1 (user D#)
  g0[1] = ldsOff;                                           // lds_addr (bytes)
  g0[2] = (unsigned)(gaddr & 0xffffffffull);                // global_addr[31:0]
  g0[3] = (unsigned)((gaddr >> 32) & 0x1ffffffull)          // global_addr[56:32]
        | (2u << 30);                                       // type=2 ("image")
  i32x8 g1;
  g1[0] = (2 << 16);                        // data_size=4B; no multicast/pad
  g1[1] = (int)(128u << 16);                // tensor_dim0=128   (bits 79:48)
  g1[2] = (int)(((BATCH * NNODE) & 0xffff) << 16);  // tensor_dim1 lo (95:80)
  g1[3] = (int)(128u << 16)                 // tile_dim0=128     (127:112)
        | (int)((BATCH * NNODE) >> 16);     // tensor_dim1 hi    (111:96)
  g1[4] = 16;                               // tile_dim1=16 rows; tile_dim2=0
  g1[5] = 128;                              // tensor_dim0_stride=128 (191:160)
  g1[6] = 0;                                // stride0 hi=0; dim1_stride lo=0
  g1[7] = 0;                                // dim1_stride hi=0
  i32x4 z4 = {0, 0, 0, 0};                  // D# groups 2/3 unused (2-D tile)
  i32x8 z8 = {0, 0, 0, 0, 0, 0, 0, 0};
  __builtin_amdgcn_tensor_load_to_lds(g0, g1, z4, z4, z8, 0);   // cpol=0
}
#endif

// ---------------------------------------------------------------------------
// Kernel 1: convert decoder weights to f16, transposed [n][k] for B-fragments.
// ---------------------------------------------------------------------------
__global__ void convert_weights(const float* __restrict__ Wfc,  // [129,128]
                                const float* __restrict__ Wfc1, // [128,128]
                                const float* __restrict__ Ww,
                                const float* __restrict__ Wat,
                                _Float16* __restrict__ WfcT, _Float16* __restrict__ Wfc1T,
                                _Float16* __restrict__ WwT,  _Float16* __restrict__ WatT,
                                float* __restrict__ wlast) {
  int t = blockIdx.x * blockDim.x + threadIdx.x;
  int stride = gridDim.x * blockDim.x;
  for (int e = t; e < DIM * DIM; e += stride) {
    int n = e >> 7, k = e & 127;
    WfcT [n * DIM + k] = (_Float16)Wfc [k * DIM + n];
    Wfc1T[n * DIM + k] = (_Float16)Wfc1[k * DIM + n];
    WwT  [n * DIM + k] = (_Float16)Ww  [k * DIM + n];
    WatT [n * DIM + k] = (_Float16)Wat [k * DIM + n];
  }
  if (t < DIM) wlast[t] = Wfc[DIM * DIM + t];  // 129th row of W_fc (capacity col)
}

// ---------------------------------------------------------------------------
// Kernel 2: WMMA precompute of Kh = enc@W_k, Vh = enc@W_v, Kp = enc@W_pk (f16).
// enc is [512*101, 128] flat; 3232 strips of 16 rows; 202 WGs x 4 waves,
// exactly 4 strips per wave.  A-tiles are DMA'd by the Tensor Data Mover into
// a per-wave double buffer (issue strip i+1, s_wait_tensorcnt 1, compute i),
// overlapping the HBM/L2 traffic with the 96-WMMA chain per strip.
// ---------------------------------------------------------------------------
#define PRE_WGS     202
#define STAGE_BYTES (16 * DIM * 4)                       // one f32 tile, 8 KB
#define PRE_W_OFF   0                                    // 3 x [128][128] f16
#define PRE_ST_OFF  (3 * DIM * DIM * 2)                  // 4 waves x 2 x 8 KB
#define PRE_AH_OFF  (PRE_ST_OFF + 4 * 2 * STAGE_BYTES)   // 4 waves x [16][128] f16
#define PRE_LDS_TOT (PRE_AH_OFF + 4 * 16 * DIM * 2)      // ~176 KB

__global__ void precompute_proj(const float* __restrict__ enc,
                                const float* __restrict__ Wk,
                                const float* __restrict__ Wv,
                                const float* __restrict__ Wpk,
                                _Float16* __restrict__ khH,
                                _Float16* __restrict__ vhH,
                                _Float16* __restrict__ kpH) {
  extern __shared__ char smem[];   // dynamic LDS starts at LDS offset 0
  _Float16* WT = (_Float16*)(smem + PRE_W_OFF);
  int tid = threadIdx.x, lane = tid & 31;
  int wave = __builtin_amdgcn_readfirstlane((int)(tid >> 5));  // SGPR wave id
  int cn = lane & 15, cm0 = (lane >> 4) * 8;

  for (int e = tid; e < DIM * DIM; e += 128) {
    int n = e >> 7, k = e & 127;
    WT[0 * 16384 + n * DIM + k] = (_Float16)Wk [k * DIM + n];
    WT[1 * 16384 + n * DIM + k] = (_Float16)Wv [k * DIM + n];
    WT[2 * 16384 + n * DIM + k] = (_Float16)Wpk[k * DIM + n];
  }
  __syncthreads();

  _Float16* A = (_Float16*)(smem + PRE_AH_OFF) + wave * 16 * DIM;
  const int myBase = blockIdx.x * 4 + wave;   // strips: myBase + i*808, i=0..3

#ifdef USE_TDM
  const unsigned stage0 = PRE_ST_OFF + (unsigned)wave * 2 * STAGE_BYTES;
  // prime the pipeline with strip 0's tile
  tdm_load_tile(stage0,
                (unsigned long long)(const void*)(enc + (size_t)myBase * 16 * DIM));
#endif

  for (int i = 0; i < 4; ++i) {
    int strip = myBase + i * PRE_WGS * 4;
    size_t row0 = (size_t)strip * 16;

#ifdef USE_TDM
    if (i < 3) {                      // overlap next DMA with this strip's math
      int nstrip = myBase + (i + 1) * PRE_WGS * 4;
      tdm_load_tile(stage0 + (unsigned)((i + 1) & 1) * STAGE_BYTES,
                    (unsigned long long)(const void*)(enc + (size_t)nstrip * 16 * DIM));
      __builtin_amdgcn_s_wait_tensorcnt((short)1);   // tile i resident
    } else {
      __builtin_amdgcn_s_wait_tensorcnt((short)0);
    }
    const float* st = (const float*)(smem + stage0 + (unsigned)(i & 1) * STAGE_BYTES);
    for (int e = lane; e < 16 * DIM; e += 32) A[e] = (_Float16)st[e];
#else
    for (int e = lane; e < 16 * DIM; e += 32) A[e] = (_Float16)enc[row0 * DIM + e];
#endif

#pragma unroll
    for (int w = 0; w < 3; ++w) {
      const _Float16* Bt = WT + w * 16384;
      _Float16* outp = (w == 0) ? khH : (w == 1) ? vhH : kpH;
      for (int nt = 0; nt < 8; ++nt) {
        v8f acc = {};
#pragma unroll
        for (int kc = 0; kc < 4; ++kc)
          acc = wmma16(ldA(A, DIM, kc * 32, lane), ldB(Bt, DIM, kc * 32, nt * 16, lane), acc);
#pragma unroll
        for (int r = 0; r < 8; ++r)
          outp[(row0 + cm0 + r) * DIM + nt * 16 + cn] = (_Float16)acc[r];
      }
    }
  }
}

// ---------------------------------------------------------------------------
// Kernel 3: persistent decoder. 32 WGs x 128 threads (4 waves); each WG owns 16
// batch rows and iterates all 140 steps locally (batches are independent, so
// no global synchronization is ever needed).  All four per-step GEMMs are
// WMMA on f16 weights resident in LDS for the whole kernel.
// ---------------------------------------------------------------------------
// LDS layout (bytes)
constexpr int L_WFC   = 0;
constexpr int L_WFC1  = L_WFC   + DIM * DIM * 2;
constexpr int L_WW    = L_WFC1  + DIM * DIM * 2;
constexpr int L_WAT   = L_WW    + DIM * DIM * 2;
constexpr int L_WLAST = L_WAT   + DIM * DIM * 2;   // float[128]
constexpr int L_INP   = L_WLAST + DIM * 4;         // f16 [16][132] (concat input)
constexpr int L_POOL0 = L_INP   + 16 * 132 * 2;    // f16 [16][128]
constexpr int L_POOL1 = L_POOL0 + 16 * DIM * 2;
constexpr int L_DECH  = L_POOL1 + 16 * DIM * 2;    // f16 [16][128]
constexpr int L_Q     = L_DECH  + 16 * DIM * 2;    // f32 [16][128]
constexpr int L_GL    = L_Q     + 16 * DIM * 4;    // f16 [16][128]
constexpr int L_G2    = L_GL    + 16 * DIM * 2;    // f32 [16][128]
constexpr int L_C2    = L_G2    + 16 * DIM * 4;    // f32 [16][104]
constexpr int L_DEM   = L_C2    + 16 * 104 * 4;    // f32 [16][104]
constexpr int L_MASK  = L_DEM   + 16 * 104 * 4;    // u8  [16][104]
constexpr int L_MASK1 = L_MASK  + 16 * 104;
constexpr int L_IDX   = L_MASK1 + 16 * 104;        // int[16]
constexpr int L_DYN   = L_IDX   + 64;              // f32[16]
constexpr int L_LOGP  = L_DYN   + 64;              // f32[16]
constexpr int L_END   = L_LOGP  + 64;

__global__ void decoder_persistent(const float* __restrict__ enc,
                                   const float* __restrict__ pool,
                                   const float* __restrict__ capc,
                                   const float* __restrict__ demand,
                                   const _Float16* __restrict__ khH,
                                   const _Float16* __restrict__ vhH,
                                   const _Float16* __restrict__ kpH,
                                   const _Float16* __restrict__ WfcT,
                                   const _Float16* __restrict__ Wfc1T,
                                   const _Float16* __restrict__ WwT,
                                   const _Float16* __restrict__ WatT,
                                   const float* __restrict__ wlast,
                                   float* __restrict__ out) {
  extern __shared__ char smem[];
  _Float16* sWfc  = (_Float16*)(smem + L_WFC);
  _Float16* sWfc1 = (_Float16*)(smem + L_WFC1);
  _Float16* sWw   = (_Float16*)(smem + L_WW);
  _Float16* sWat  = (_Float16*)(smem + L_WAT);
  float*    sWl   = (float*)   (smem + L_WLAST);
  _Float16* sInp  = (_Float16*)(smem + L_INP);
  _Float16* sP0   = (_Float16*)(smem + L_POOL0);
  _Float16* sP1   = (_Float16*)(smem + L_POOL1);
  _Float16* sDecH = (_Float16*)(smem + L_DECH);
  float*    sQ    = (float*)   (smem + L_Q);
  _Float16* sGl   = (_Float16*)(smem + L_GL);
  float*    sG2   = (float*)   (smem + L_G2);
  float*    sC2   = (float*)   (smem + L_C2);
  float*    sDem  = (float*)   (smem + L_DEM);
  unsigned char* sM  = (unsigned char*)(smem + L_MASK);
  unsigned char* sM1 = (unsigned char*)(smem + L_MASK1);
  int*   sIdx  = (int*)  (smem + L_IDX);
  float* sDyn  = (float*)(smem + L_DYN);
  float* sLogp = (float*)(smem + L_LOGP);

  int tid = threadIdx.x, lane = tid & 31, wave = tid >> 5;
  int cn = lane & 15, cm0 = (lane >> 4) * 8;
  size_t gb0 = (size_t)blockIdx.x * 16;
  float cap0 = capc[0];

  // ---- init: weights -> LDS, per-batch state
  for (int e = tid; e < DIM * DIM; e += 128) {
    sWfc[e] = WfcT[e]; sWfc1[e] = Wfc1T[e]; sWw[e] = WwT[e]; sWat[e] = WatT[e];
  }
  if (tid < DIM) sWl[tid] = wlast[tid];
  for (int e = tid; e < 16 * DIM; e += 128) {
    int b = e >> 7, k = e & 127;
    sP0[b * DIM + k]  = (_Float16)pool[(gb0 + b) * DIM + k];
    sInp[b * 132 + k] = (_Float16)enc[((gb0 + b) * NNODE + 0) * DIM + k];
  }
  for (int e = tid; e < 16 * NNODE; e += 128) {
    int b = e / NNODE, n = e % NNODE;
    sDem[b * 104 + n] = demand[(gb0 + b) * NNODE + n];
  }
  __syncthreads();
  if (tid < 16) {
    int b = tid;
    float capb = capc[gb0 + b];
    sDyn[b] = capb; sLogp[b] = 0.f;
    sInp[b * 132 + DIM] = (_Float16)capb;
    for (int n = 0; n < NNODE; ++n) {           // update_mask(0, cap, idx=0)
      unsigned char m1 = (n == 0);
      sM1[b * 104 + n] = m1;
      sM[b * 104 + n]  = (unsigned char)(m1 || (sDem[b * 104 + n] > capb));
    }
  }
  __syncthreads();

  // ---- 140 sequential decode steps (LDS barriers only)
  for (int s = 0; s < NSTEP; ++s) {
    const _Float16* pc = (s & 1) ? sP1 : sP0;   // pool_c (ping)
    _Float16*       pn = (s & 1) ? sP0 : sP1;   // pool_n (pong, next pool_c)

    // dec = [inp|cap]@W_fc + pool@W_fc1 ; also emit pool_n   (WMMA, 2 tiles/wave)
    for (int t = wave; t < 8; t += 4) {
      v8f acc = {};
#pragma unroll
      for (int kc = 0; kc < 4; ++kc)
        acc = wmma16(ldA(pc, DIM, kc * 32, lane), ldB(sWfc1, DIM, kc * 32, t * 16, lane), acc);
#pragma unroll
      for (int r = 0; r < 8; ++r)               // pool_n for next step
        pn[(cm0 + r) * DIM + t * 16 + cn] = (_Float16)acc[r];
#pragma unroll
      for (int kc = 0; kc < 4; ++kc)
        acc = wmma16(ldA(sInp, 132, kc * 32, lane), ldB(sWfc, DIM, kc * 32, t * 16, lane), acc);
#pragma unroll
      for (int r = 0; r < 8; ++r) {             // rank-1 capacity column (K=129th)
        float d = acc[r] + sDyn[cm0 + r] * sWl[t * 16 + cn];
        sDecH[(cm0 + r) * DIM + t * 16 + cn] = (_Float16)d;
      }
    }
    __syncthreads();

    // Q = dec @ W_w   (WMMA)
    for (int t = wave; t < 8; t += 4) {
      v8f acc = {};
#pragma unroll
      for (int kc = 0; kc < 4; ++kc)
        acc = wmma16(ldA(sDecH, DIM, kc * 32, lane), ldB(sWw, DIM, kc * 32, t * 16, lane), acc);
#pragma unroll
      for (int r = 0; r < 8; ++r) sQ[(cm0 + r) * DIM + t * 16 + cn] = acc[r];
    }
    __syncthreads();

    // masked MHA: one thread per (batch,head); online softmax over 101 nodes
    {
      int b = tid >> 3, h = tid & 7;
      const _Float16* kb_ = khH + (gb0 + b) * NNODE * DIM + h * HD;
      const _Float16* vb_ = vhH + (gb0 + b) * NNODE * DIM + h * HD;
      const unsigned char* mk = sM + b * 104;
      float q[HD];
#pragma unroll
      for (int d = 0; d < HD; ++d) q[d] = sQ[b * DIM + h * HD + d];
      float mmax = -3.0e38f, ssum = 0.f, acc[HD];
#pragma unroll
      for (int d = 0; d < HD; ++d) acc[d] = 0.f;
      for (int n = 0; n < NNODE; ++n) {
        if (mk[n]) continue;
        const _Float16* kp = kb_ + (size_t)n * DIM;
        const _Float16* vp = vb_ + (size_t)n * DIM;
        __builtin_prefetch((const void*)(kp + DIM), 0, 0);  // global_prefetch_b8
        float u = 0.f;
#pragma unroll
        for (int d = 0; d < HD; ++d) u += q[d] * (float)kp[d];
        u *= 0.25f;  // 1/sqrt(16)
        float w;
        if (u > mmax) {
          float sc = __expf(mmax - u);
          ssum *= sc;
#pragma unroll
          for (int d = 0; d < HD; ++d) acc[d] *= sc;
          mmax = u; w = 1.f;
        } else {
          w = __expf(u - mmax);
        }
        ssum += w;
#pragma unroll
        for (int d = 0; d < HD; ++d) acc[d] += w * (float)vp[d];
      }
      float inv = 1.f / ssum;
#pragma unroll
      for (int d = 0; d < HD; ++d)
        sGl[b * DIM + h * HD + d] = (_Float16)(acc[d] * inv);
    }
    __syncthreads();

    // glimpse2 = glimpse @ W_attnfc   (WMMA)
    for (int t = wave; t < 8; t += 4) {
      v8f acc = {};
#pragma unroll
      for (int kc = 0; kc < 4; ++kc)
        acc = wmma16(ldA(sGl, DIM, kc * 32, lane), ldB(sWat, DIM, kc * 32, t * 16, lane), acc);
#pragma unroll
      for (int r = 0; r < 8; ++r) sG2[(cm0 + r) * DIM + t * 16 + cn] = acc[r];
    }
    __syncthreads();

    // pointer scores: comp2[b,n] = g2[b,:]·Kp[b,n,:] / sqrt(128)  (batched GEMV)
    for (int e = tid; e < 16 * NNODE; e += 128) {
      int b = e / NNODE, n = e % NNODE;
      const _Float16* kp = kpH + ((gb0 + b) * NNODE + n) * DIM;
      float sum = 0.f;
#pragma unroll 16
      for (int h = 0; h < DIM; ++h) sum += sG2[b * DIM + h] * (float)kp[h];
      sC2[b * 104 + n] = sum * 0.08838834764831844f;
    }
    __syncthreads();

    // per-batch: logits, argmax, log_p, state/mask update (thread b owns batch b)
    if (tid < 16) {
      int b = tid;
      float best = -3.0e38f; int bidx = 0;
      for (int n = 0; n < NNODE; ++n) {
        float l = sM[b * 104 + n] ? -3.0e38f : 10.f * tanhf(sC2[b * 104 + n]);
        sC2[b * 104 + n] = l;
        if (l > best) { best = l; bidx = n; }   // first-max == jnp.argmax
      }
      float se = 0.f;
      for (int n = 0; n < NNODE; ++n)
        if (!sM[b * 104 + n]) se += __expf(sC2[b * 104 + n] - best);
      float logp = -__logf(se);                 // logit_max - logsumexp
      int cnt = 0;                              // is_done from PRE-update mask1
      for (int n = 1; n < NNODE; ++n) cnt += sM1[b * 104 + n];
      if (cnt >= NNODE - 1) logp = 0.f;
      sLogp[b] += logp;
      out[(gb0 + b) * NSTEP + s] = (float)bidx;

      float dyn = (bidx == 0) ? cap0 : (sDyn[b] - sDem[b * 104 + bidx]);
      sDyn[b] = dyn;
      sM1[b * 104 + bidx] = 1;
      sM1[b * 104 + 0]    = (bidx == 0) ? 1 : 0;  // .set(go_depot)
      int all = 1;
      for (int n = 0; n < NNODE; ++n) {
        int mm = sM1[b * 104 + n] || (sDem[b * 104 + n] > dyn);
        sM[b * 104 + n] = (unsigned char)mm;
        all &= mm;
      }
      if (all) sM[b * 104 + 0] = 0;
      sIdx[b] = bidx;
    }
    __syncthreads();

    // gather next input embedding + capacity into concat buffer
    for (int e = tid; e < 16 * DIM; e += 128) {
      int b = e >> 7, k = e & 127;
      sInp[b * 132 + k] = (_Float16)enc[((gb0 + b) * NNODE + sIdx[b]) * DIM + k];
    }
    if (tid < 16) sInp[tid * 132 + DIM] = (_Float16)sDyn[tid];
    __syncthreads();
  }

  if (tid < 16) out[(size_t)BATCH * NSTEP + gb0 + tid] = sLogp[tid];
}

// ---------------------------------------------------------------------------
extern "C" void kernel_launch(void* const* d_in, const int* in_sizes, int n_in,
                              void* d_out, int out_size, void* d_ws, size_t ws_size,
                              hipStream_t stream) {
  (void)in_sizes; (void)n_in; (void)out_size; (void)ws_size;
  const float* enc    = (const float*)d_in[0];
  const float* pool   = (const float*)d_in[1];
  const float* capc   = (const float*)d_in[3];
  const float* demand = (const float*)d_in[4];
  const float* Wfc    = (const float*)d_in[7];
  const float* Wfc1   = (const float*)d_in[8];
  const float* Ww     = (const float*)d_in[9];
  const float* Wk     = (const float*)d_in[10];
  const float* Wv     = (const float*)d_in[11];
  const float* Wat    = (const float*)d_in[12];
  const float* Wpk    = (const float*)d_in[13];
  float* out = (float*)d_out;

  char* ws = (char*)d_ws;
  size_t off = 0;
  auto carve = [&](size_t bytes) -> char* {
    char* p = ws + off; off += (bytes + 255) & ~(size_t)255; return p;
  };
  _Float16* khH   = (_Float16*)carve((size_t)BATCH * NNODE * DIM * 2);
  _Float16* vhH   = (_Float16*)carve((size_t)BATCH * NNODE * DIM * 2);
  _Float16* kpH   = (_Float16*)carve((size_t)BATCH * NNODE * DIM * 2);
  _Float16* WfcT  = (_Float16*)carve(DIM * DIM * 2);
  _Float16* Wfc1T = (_Float16*)carve(DIM * DIM * 2);
  _Float16* WwT   = (_Float16*)carve(DIM * DIM * 2);
  _Float16* WatT  = (_Float16*)carve(DIM * DIM * 2);
  float*    wlast = (float*)   carve(DIM * 4);

  convert_weights<<<64, 128, 0, stream>>>(Wfc, Wfc1, Ww, Wat,
                                          WfcT, Wfc1T, WwT, WatT, wlast);

  precompute_proj<<<PRE_WGS, 128, PRE_LDS_TOT, stream>>>(enc, Wk, Wv, Wpk,
                                                         khH, vhH, kpH);

  decoder_persistent<<<BATCH / 16, 128, L_END, stream>>>(
      enc, pool, capc, demand, khH, vhH, kpH,
      WfcT, Wfc1T, WwT, WatT, wlast, out);
}
